// ContrastiveLoss_21019569946956
// MI455X (gfx1250) — matrix operand order repaired and checked by reading, not compiled, not run on hardware
//
#include <hip/hip_runtime.h>

// ---------------------------------------------------------------------------
// ContrastiveLoss for MI455X (gfx1250, wave32, WMMA + async LDS staging)
//   text, visual : [B=512, T=196, D=256] fp32
//   loss         : scalar fp32
//
// Pipeline:
//   1) normalize_kernel : per-token L2 normalize, fp32 -> bf16, into d_ws
//   2) tok_sim_kernel   : fused (per-t 512x512 GEMM_K=256) + online softmax
//                         pooling over t. v_wmma_f32_16x16x32_bf16, with
//                         global_load_async_to_lds_b128 double-buffered
//                         staging (ASYNCcnt-tracked, VGPR-bypassing).
//   3) lse_diag_kernel  : row/col logsumexp of logits + diagonal terms
//   4) final_loss_kernel: deterministic reduction -> d_out[0]
// ---------------------------------------------------------------------------

typedef __bf16 bf16x4 __attribute__((ext_vector_type(4)));
typedef __bf16 bf16x8 __attribute__((ext_vector_type(8)));
typedef __bf16 v16bf  __attribute__((ext_vector_type(16)));
typedef float  v8f    __attribute__((ext_vector_type(8)));

#define BSZ   512
#define TTOK  196
#define DDIM  256
#define LSTR  264   // padded LDS row stride in bf16 elems (528 B -> conflict-free b128)
#define INV_TEMP (1.0f / 0.07f)

union FragU { bf16x8 h[2]; v16bf v; };

static __device__ __forceinline__ v16bf frag2(const unsigned short* p0,
                                              const unsigned short* p1) {
  FragU u;
  u.h[0] = *reinterpret_cast<const bf16x8*>(p0);
  u.h[1] = *reinterpret_cast<const bf16x8*>(p1);
  return u.v;
}

// Per-lane async copy: 16 B global -> LDS, tracked by ASYNCcnt, bypasses VGPRs.
static __device__ __forceinline__ void async_ld_b128(unsigned lds_off,
                                                     const void* gaddr) {
  asm volatile("global_load_async_to_lds_b128 %0, %1, off"
               :: "v"(lds_off), "v"(gaddr)
               : "memory");
}

// --------------------------------------------------------------------------
// 1) Per-token L2 normalize + convert to bf16.  One wave per (b,t) row.
//    grid = (nrows/8, 2), block = 256
// --------------------------------------------------------------------------
__global__ __launch_bounds__(256) void normalize_kernel(
    const float* __restrict__ tSrc, const float* __restrict__ vSrc,
    unsigned short* __restrict__ tDst, unsigned short* __restrict__ vDst) {
  const float* src = blockIdx.y ? vSrc : tSrc;
  unsigned short* dst = blockIdx.y ? vDst : tDst;

  const int row  = blockIdx.x * 8 + (threadIdx.x >> 5);
  const int lane = threadIdx.x & 31;

  const float4* s4 = reinterpret_cast<const float4*>(src + (size_t)row * DDIM);
  float4 a = s4[lane];
  float4 b = s4[lane + 32];

  float ss = a.x * a.x + a.y * a.y + a.z * a.z + a.w * a.w
           + b.x * b.x + b.y * b.y + b.z * b.z + b.w * b.w;
#pragma unroll
  for (int off = 16; off > 0; off >>= 1) ss += __shfl_xor(ss, off, 32);

  const float scale = rsqrtf(fmaxf(ss, 1e-24f));

  bf16x4 oa, ob;
  oa[0] = (__bf16)(a.x * scale); oa[1] = (__bf16)(a.y * scale);
  oa[2] = (__bf16)(a.z * scale); oa[3] = (__bf16)(a.w * scale);
  ob[0] = (__bf16)(b.x * scale); ob[1] = (__bf16)(b.y * scale);
  ob[2] = (__bf16)(b.z * scale); ob[3] = (__bf16)(b.w * scale);

  unsigned short* drow = dst + (size_t)row * DDIM;
  *reinterpret_cast<bf16x4*>(drow + lane * 4)       = oa;
  *reinterpret_cast<bf16x4*>(drow + 128 + lane * 4) = ob;
}

// --------------------------------------------------------------------------
// 2) Fused token-GEMM + softmax pooling over t.
//    Output tile per WG: 32 (b rows) x 64 (c cols). grid = (8, 16), block 256.
//    8 waves, each owns one 16x16 WMMA tile. Double-buffered async staging.
//    Online pooling with fixed shift (cos <= 1):  s += e^(c-1), u += c*e^(c-1)
// --------------------------------------------------------------------------
__global__ __launch_bounds__(256) void tok_sim_kernel(
    const unsigned short* __restrict__ tn,   // normalized text  bf16 [B,T,D]
    const unsigned short* __restrict__ vn,   // normalized visual bf16 [B,T,D]
    float* __restrict__ logits) {            // [B,B] fp32 (already / TEMP)
  __shared__ __align__(16) unsigned short ldsA[2][32 * LSTR];
  __shared__ __align__(16) unsigned short ldsB[2][64 * LSTR];

  const int tid      = threadIdx.x;
  const int lane     = tid & 31;
  const int wave     = tid >> 5;
  const int wy       = wave >> 2;       // 0..1  (16-row band)
  const int wx       = wave & 3;        // 0..3  (16-col band)
  const int rowBase  = wy * 16;
  const int colBase  = wx * 16;
  const int laneLow  = lane & 15;
  const int laneHalf = lane >> 4;

  const int b0 = blockIdx.y * 32;
  const int c0 = blockIdx.x * 64;

  float sAcc[8], uAcc[8];
#pragma unroll
  for (int e = 0; e < 8; ++e) { sAcc[e] = 0.0f; uAcc[e] = 0.0f; }

  const uint4* tG = reinterpret_cast<const uint4*>(tn);
  const uint4* vG = reinterpret_cast<const uint4*>(vn);

  // 32-bit LDS offsets for the async destinations (low 32 bits of the
  // generic pointer are the LDS byte offset; shared aperture is in the
  // high half).
  const unsigned ldsA_base[2] = {(unsigned)(size_t)(void*)&ldsA[0][0],
                                 (unsigned)(size_t)(void*)&ldsA[1][0]};
  const unsigned ldsB_base[2] = {(unsigned)(size_t)(void*)&ldsB[0][0],
                                 (unsigned)(size_t)(void*)&ldsB[1][0]};

  // Per-thread staging coordinates (16 B granules; rows of 512 B coalesced)
  const int rA = tid >> 5, cA = tid & 31;   // A: 32 rows x 32 uint4, 4 passes

  // Issue the 12 async b128 copies for token t into buffer `buf`.
  auto stage = [&](int t, int buf) {
#pragma unroll
    for (int p = 0; p < 4; ++p) {
      const int r = rA + p * 8;
      async_ld_b128(ldsA_base[buf] + (unsigned)((r * 33 + cA) * 16),
                    (const void*)(tG + ((size_t)(b0 + r) * TTOK + t) * 32 + cA));
    }
#pragma unroll
    for (int p = 0; p < 8; ++p) {
      const int r = rA + p * 8;
      async_ld_b128(ldsB_base[buf] + (unsigned)((r * 33 + cA) * 16),
                    (const void*)(vG + ((size_t)(c0 + r) * TTOK + t) * 32 + cA));
    }
  };

  // Prologue: prefetch t = 0
  stage(0, 0);

  for (int t = 0; t < TTOK; ++t) {
    const int buf = t & 1;

    // Prefetch next token's slabs into the other buffer while computing.
    if (t + 1 < TTOK) {
      stage(t + 1, buf ^ 1);
      // 12 newest ops are the prefetch; in-order completion => ASYNCcnt<=12
      // guarantees the current buffer's 12 copies have landed.
      asm volatile("s_wait_asynccnt 12" ::: "memory");
    } else {
      asm volatile("s_wait_asynccnt 0" ::: "memory");
    }
    __syncthreads();   // all waves' current-buffer copies are complete

    const unsigned short* aRow = &ldsA[buf][(rowBase + laneLow) * LSTR];
    const unsigned short* bRow = &ldsB[buf][(colBase + laneLow) * LSTR];

    v8f acc = {0.f, 0.f, 0.f, 0.f, 0.f, 0.f, 0.f, 0.f};
#pragma unroll
    for (int kb = 0; kb < DDIM; kb += 32) {
      // A 16x32 bf16 fragment: lane(<16) -> row M=lane, K = kb+{0..7,16..23};
      // lanes 16..31 -> K = kb+{8..15, 24..31}
      v16bf af = frag2(aRow + kb + laneHalf * 8,
                       aRow + kb + 16 + laneHalf * 8);
      // B 32x16 bf16 fragment: lane(<16) -> col N=lane, K = kb+{0..7,16..23};
      // lanes 16..31 -> K = kb+{8..15, 24..31}
      v16bf bf = frag2(bRow + kb + laneHalf * 8,
                       bRow + kb + 16 + laneHalf * 8);
      acc = __builtin_amdgcn_wmma_f32_16x16x32_bf16(
          false, af, false, bf, (short)0, acc, false, false);
    }

    // Online softmax pooling over t (shift fixed at 1.0: cos <= 1)
#pragma unroll
    for (int e = 0; e < 8; ++e) {
      const float cv = acc[e];
      const float w  = __expf(cv - 1.0f);
      sAcc[e] += w;
      uAcc[e]  = fmaf(cv, w, uAcc[e]);
    }

    // All reads of `buf` done before iteration t+1 overwrites it via async.
    __syncthreads();
  }

  // C/D layout: VGPR e -> M = e (lanes 0-15) / e+8 (lanes 16-31), N = laneLow
  const int gcol = c0 + colBase + laneLow;
#pragma unroll
  for (int e = 0; e < 8; ++e) {
    const int grow = b0 + rowBase + laneHalf * 8 + e;
    logits[(size_t)grow * BSZ + gcol] = (uAcc[e] / sAcc[e]) * INV_TEMP;
  }
}

// --------------------------------------------------------------------------
// 3) Per-index b: rowLSE(b) + colLSE(b) - 2*diag(b).  grid = 512, block 256.
// --------------------------------------------------------------------------
__global__ __launch_bounds__(256) void lse_diag_kernel(
    const float* __restrict__ logits, float* __restrict__ partial) {
  __shared__ float red[256];
  const int b = blockIdx.x;
  const int tid = threadIdx.x;

  const float r0 = logits[(size_t)b * BSZ + tid];
  const float r1 = logits[(size_t)b * BSZ + tid + 256];
  const float q0 = logits[(size_t)tid * BSZ + b];
  const float q1 = logits[(size_t)(tid + 256) * BSZ + b];

  // row max
  red[tid] = fmaxf(r0, r1);
  __syncthreads();
  for (int s = 128; s > 0; s >>= 1) {
    if (tid < s) red[tid] = fmaxf(red[tid], red[tid + s]);
    __syncthreads();
  }
  const float rowMax = red[0];
  __syncthreads();
  // row sum-exp
  red[tid] = __expf(r0 - rowMax) + __expf(r1 - rowMax);
  __syncthreads();
  for (int s = 128; s > 0; s >>= 1) {
    if (tid < s) red[tid] += red[tid + s];
    __syncthreads();
  }
  const float rowLSE = rowMax + __logf(red[0]);
  __syncthreads();
  // col max
  red[tid] = fmaxf(q0, q1);
  __syncthreads();
  for (int s = 128; s > 0; s >>= 1) {
    if (tid < s) red[tid] = fmaxf(red[tid], red[tid + s]);
    __syncthreads();
  }
  const float colMax = red[0];
  __syncthreads();
  // col sum-exp
  red[tid] = __expf(q0 - colMax) + __expf(q1 - colMax);
  __syncthreads();
  for (int s = 128; s > 0; s >>= 1) {
    if (tid < s) red[tid] += red[tid + s];
    __syncthreads();
  }
  const float colLSE = colMax + __logf(red[0]);

  if (tid == 0) {
    const float diag = logits[(size_t)b * BSZ + b];
    partial[b] = rowLSE + colLSE - 2.0f * diag;
  }
}

// --------------------------------------------------------------------------
// 4) loss = (1/(2B)) * sum_b partial[b]
// --------------------------------------------------------------------------
__global__ __launch_bounds__(256) void final_loss_kernel(
    const float* __restrict__ partial, float* __restrict__ out) {
  __shared__ float red[256];
  const int tid = threadIdx.x;
  red[tid] = partial[tid] + partial[tid + 256];
  __syncthreads();
  for (int s = 128; s > 0; s >>= 1) {
    if (tid < s) red[tid] += red[tid + s];
    __syncthreads();
  }
  if (tid == 0) out[0] = red[0] * (1.0f / (2.0f * (float)BSZ));
}

// --------------------------------------------------------------------------
extern "C" void kernel_launch(void* const* d_in, const int* in_sizes, int n_in,
                              void* d_out, int out_size, void* d_ws,
                              size_t ws_size, hipStream_t stream) {
  const float* text   = (const float*)d_in[0];
  const float* visual = (const float*)d_in[1];
  float* out = (float*)d_out;

  const size_t nTok = (size_t)BSZ * TTOK * DDIM;  // 25,690,112 elems
  char* ws = (char*)d_ws;
  unsigned short* tN = (unsigned short*)ws;                   // bf16 [B,T,D]
  unsigned short* vN = tN + nTok;                             // bf16 [B,T,D]
  float* logits  = (float*)(ws + 2 * nTok * sizeof(unsigned short)); // [B,B]
  float* partial = logits + (size_t)BSZ * BSZ;                // [B]

  // 1) normalize + convert (both tensors via grid.y)
  normalize_kernel<<<dim3((BSZ * TTOK) / 8, 2), 256, 0, stream>>>(
      text, visual, tN, vN);

  // 2) fused per-token GEMM + softmax pooling -> logits (already / TEMP)
  tok_sim_kernel<<<dim3(BSZ / 64, BSZ / 32), 256, 0, stream>>>(tN, vN, logits);

  // 3) per-row/col logsumexp + diag
  lse_diag_kernel<<<BSZ, 256, 0, stream>>>(logits, partial);

  // 4) final scalar
  final_loss_kernel<<<1, 256, 0, stream>>>(partial, out);
}
